// SoftmaxV_85624468013185
// MI455X (gfx1250) — matrix operand order repaired and checked by reading, not compiled, not run on hardware
//
#include <hip/hip_runtime.h>

typedef __attribute__((ext_vector_type(16))) _Float16 v16h;
typedef __attribute__((ext_vector_type(8)))  _Float16 v8h;
typedef __attribute__((ext_vector_type(2)))  _Float16 h2;
typedef __attribute__((ext_vector_type(8)))  float    v8f;
typedef __attribute__((ext_vector_type(4)))  float    v4f;

namespace {
constexpr int kS   = 2048;  // sequence length (keys and queries)
constexpr int kD   = 64;    // head dim
constexpr int kKC  = 32;    // keys per chunk (= WMMA K for f16)
constexpr int kVW  = 40;    // padded vt row stride in halves (80B: 16B-aligned,
                            // row*20 words mod 64 distinct for rows 0..15 -> no bank conflicts)
}

// One wave handles a 16-query tile via online softmax + WMMA PV accumulate.
// 8 waves / block share one (b,h); the block double-buffers each 32x64 V chunk
// in LDS (transposed, f16) so B-fragments are two conflict-free ds_load_b128.
__global__ __launch_bounds__(256)
void softmaxv_flash_wmma(const float* __restrict__ scores,
                         const float* __restrict__ vglob,
                         float* __restrict__ out,
                         int bh_count)
{
    __shared__ _Float16 vt[2][kD * kVW];   // vt[buf][n*kVW + k], 2 x 5 KB

    const int tid  = threadIdx.x;
    const int wave = tid >> 5;          // 0..7
    const int lane = tid & 31;
    const int hgrp = lane >> 4;         // 0: lanes 0-15, 1: lanes 16-31
    const int l16  = lane & 15;

    const int bh   = blockIdx.x >> 4;   // (b*H + h)
    const int qblk = blockIdx.x & 15;   // 16 blocks per bh
    const int q0   = (qblk * 8 + wave) * 16;
    if (bh >= bh_count) return;

    const float* srow = scores + ((size_t)bh * kS + (size_t)(q0 + l16)) * (size_t)kS;
    const float* vbh  = vglob  + (size_t)bh * kS * kD;

    // Staging assignment: thread reads rows (k, k+1) cols nn..nn+3 of the V
    // chunk (fully coalesced 256B row segments) and writes 4 packed b32 pairs.
    const int kk = (tid >> 4) * 2;      // even k in 0..30
    const int nn = (tid & 15) * 4;      // col base, multiple of 4

    // Online-softmax state for this lane's A-row (row = l16; lanes l16 and
    // l16+16 carry identical stats after the xor-16 merge each chunk).
    float mrow = -__builtin_inff();
    float lrow = 0.0f;
    v8f acc[4];
    #pragma unroll
    for (int t = 0; t < 4; ++t) acc[t] = (v8f){};

    const int kb2 = hgrp * 8;           // A-layout K base within chunk for this lane

    // ---- preload chunk 0 of V into buffer 0 ----
    {
        const float* vp = vbh + (size_t)kk * kD + nn;
        v4f r0 = *(const v4f*)(vp);
        v4f r1 = *(const v4f*)(vp + kD);
        #pragma unroll
        for (int j = 0; j < 4; ++j) {
            h2 pk = { (_Float16)r0[j], (_Float16)r1[j] };
            *(h2*)(&vt[0][(nn + j) * kVW + kk]) = pk;
        }
    }
    __syncthreads();

    for (int ic = 0; ic < kS / kKC; ++ic) {
        const int k0  = ic * kKC;
        const int cur = ic & 1;
        const bool more = (k0 + kKC) < kS;

        // ---- issue next V chunk global loads early (hide under compute) ----
        v4f r0 = {}, r1 = {};
        if (more) {
            const float* vp = vbh + (size_t)(k0 + kKC + kk) * kD + nn;
            r0 = *(const v4f*)(vp);
            r1 = *(const v4f*)(vp + kD);
        }

        // ---- B fragments: issue all 8 ds_load_b128 up front, one wait,
        //      then 4 back-to-back WMMAs (breaks load->wait->wmma chain) ----
        v16h bfr[4];
        #pragma unroll
        for (int t = 0; t < 4; ++t) {
            // B 32x16 f16 layout: lane = col l16; e=0..7 -> K=8*hgrp+e, e=8..15 -> K=16+8*hgrp+e-8
            const _Float16* bp = &vt[cur][(t * 16 + l16) * kVW + hgrp * 8];
            v8h b0 = *(const v8h*)(bp);
            v8h b1 = *(const v8h*)(bp + 16);
            #pragma unroll
            for (int i = 0; i < 8; ++i) { bfr[t][i] = b0[i]; bfr[t][8 + i] = b1[i]; }
        }

        // ---- A-fragment score loads (WMMA f16 16x32 A layout) ----
        // e=0..7  -> K = kb2 + e        (b128 at +0, +4)
        // e=8..15 -> K = kb2 + 16 + e-8 (b128 at +16, +20)
        const float* sp = srow + k0 + kb2;
        v4f s0 = *(const v4f*)(sp + 0);
        v4f s1 = *(const v4f*)(sp + 4);
        v4f s2 = *(const v4f*)(sp + 16);
        v4f s3 = *(const v4f*)(sp + 20);
        if (more) __builtin_prefetch(sp + kKC, 0, 0);

        float sv[16];
        #pragma unroll
        for (int j = 0; j < 4; ++j) {
            sv[j] = s0[j]; sv[4 + j] = s1[j]; sv[8 + j] = s2[j]; sv[12 + j] = s3[j];
        }

        // chunk max for this row (merge the two half-lanes of the row)
        float cmax = sv[0];
        #pragma unroll
        for (int e = 1; e < 16; ++e) cmax = fmaxf(cmax, sv[e]);
        cmax = fmaxf(cmax, __shfl_xor(cmax, 16, 32));

        const float mold = mrow;
        const float mnew = fmaxf(mold, cmax);
        mrow = mnew;

        // Rescale only when some row's max actually grew (wave-uniform branch,
        // EXEC stays all-ones). Rows that didn't grow get scl = exp(0) = 1.
        if (__ballot(cmax > mold) != 0ULL) {
            const float scl = __expf(mold - mnew);   // first chunk: exp(-inf)=0
            lrow *= scl;
            #pragma unroll
            for (int j = 0; j < 8; ++j) {
                // C layout: VGPR j holds row j + 8*hgrp; lane r carries row r's stats
                float sj = __shfl(scl, (hgrp << 3) + j, 32);
                acc[0][j] *= sj; acc[1][j] *= sj; acc[2][j] *= sj; acc[3][j] *= sj;
            }
        }

        float psum = 0.0f;
        v16h afrag;
        #pragma unroll
        for (int e = 0; e < 16; ++e) {
            float p = __expf(sv[e] - mnew);
            psum += p;
            afrag[e] = (_Float16)p;
        }
        psum += __shfl_xor(psum, 16, 32);
        lrow += psum;

        // ---- PV accumulate: 4 x v_wmma_f32_16x16x32_f16 over D slabs ----
        #pragma unroll
        for (int t = 0; t < 4; ++t) {
            acc[t] = __builtin_amdgcn_wmma_f32_16x16x32_f16(
                false, afrag, false, bfr[t], (short)0, acc[t], false, false);
        }

        // ---- stage next V chunk into the other buffer (packed b32 stores) ----
        if (more) {
            #pragma unroll
            for (int j = 0; j < 4; ++j) {
                h2 pk = { (_Float16)r0[j], (_Float16)r1[j] };
                *(h2*)(&vt[cur ^ 1][(nn + j) * kVW + kk]) = pk;
            }
        }
        __syncthreads();   // single barrier per chunk (double-buffered)
    }

    // ---- normalize and store (C layout: VGPR j -> row j + 8*hgrp, col l16+16t) ----
    const float linv = 1.0f / lrow;
    #pragma unroll
    for (int j = 0; j < 8; ++j) {
        const int row = (hgrp << 3) + j;
        const float lj = __shfl(linv, row, 32);
        float* orow = out + ((size_t)bh * kS + (size_t)(q0 + row)) * (size_t)kD;
        #pragma unroll
        for (int t = 0; t < 4; ++t) {
            orow[t * 16 + l16] = acc[t][j] * lj;
        }
    }
}

extern "C" void kernel_launch(void* const* d_in, const int* in_sizes, int n_in,
                              void* d_out, int out_size, void* d_ws, size_t ws_size,
                              hipStream_t stream) {
    const float* scores = (const float*)d_in[0];   // [B,H,S,S] f32
    const float* v      = (const float*)d_in[1];   // [B,H,S,D] f32
    float* out          = (float*)d_out;           // [B,H,S,D] f32

    const int bh_count = in_sizes[1] / (kS * kD);  // B*H = 32
    dim3 block(256);                               // 8 wave32s
    dim3 grid(bh_count * 16);                      // 16 blocks per (b,h)
    hipLaunchKernelGGL(softmaxv_flash_wmma, grid, block, 0, stream,
                       scores, v, out, bh_count);
}
